// CombinedModel_wGCN_variable_89747636617338
// MI455X (gfx1250) — matrix operand order repaired and checked by reading, not compiled, not run on hardware
//
#include <hip/hip_runtime.h>

#define NN     128   // nodes per graph
#define UNITS  128
#define HIDDEN 6
#define KIN    32    // input K (8 feat + 8 emb, zero-padded to 32)

typedef __bf16 bf16_t;
typedef bf16_t bf16x4 __attribute__((ext_vector_type(4)));
typedef bf16_t bf16x8 __attribute__((ext_vector_type(8)));
typedef bf16_t v16bf  __attribute__((ext_vector_type(16)));
typedef float  v4f    __attribute__((ext_vector_type(4)));
typedef float  v8f    __attribute__((ext_vector_type(8)));

// ---------------- WMMA fragment helpers (wave32, 16x16x32 bf16) ----------------
// A-matrix 16x32 bf16 per-lane layout (ISA 7.12.2):
//   lane<16: row=lane, K = {0..7} then {16..23};  lane>=16: row=lane-16, K={8..15},{24..31}
__device__ __forceinline__ v16bf load_a_frag(const bf16_t* __restrict__ tile, int stride) {
    const int lane = threadIdx.x & 31;
    const int row  = lane & 15;
    const int kb   = (lane >> 4) << 3;      // 0 or 8
    const bf16_t* p = tile + row * stride + kb;
    bf16x8 lo = *(const bf16x8*)p;          // K = kb .. kb+7
    bf16x8 hi = *(const bf16x8*)(p + 16);   // K = 16+kb .. 16+kb+7
    return __builtin_shufflevector(lo, hi, 0, 1, 2, 3, 4, 5, 6, 7,
                                            8, 9, 10, 11, 12, 13, 14, 15);
}

// B-matrix 32x16 bf16 from a TRANSPOSED tile BT[n][k] (row = N, col = K):
//   lane<16: n=lane, K={0..7},{16..23};  lane>=16: n=lane-16, K={8..15},{24..31}
//   -> two 16B-aligned bf16x8 loads per lane.
__device__ __forceinline__ v16bf load_bt_frag(const bf16_t* __restrict__ tileT, int stride) {
    const int lane = threadIdx.x & 31;
    const int col  = lane & 15;
    const int kb   = (lane >> 4) << 3;      // 0 or 8
    const bf16_t* p = tileT + col * stride + kb;
    bf16x8 lo = *(const bf16x8*)p;
    bf16x8 hi = *(const bf16x8*)(p + 16);
    return __builtin_shufflevector(lo, hi, 0, 1, 2, 3, 4, 5, 6, 7,
                                            8, 9, 10, 11, 12, 13, 14, 15);
}

__device__ __forceinline__ v8f wmma_bf16(v16bf a, v16bf b, v8f c) {
    return __builtin_amdgcn_wmma_f32_16x16x32_bf16(false, a, false, b, (short)0, c, false, false);
}

// Vector epilogues: keep everything in vector form so clang can emit
// packed v_pk_add_f32 / v_pk_max / paired v_cvt_pk_bf16_f32 without moves.
__device__ __forceinline__ bf16x8 bias_cvt(v8f acc, float bias) {
    v8f bv = {bias, bias, bias, bias, bias, bias, bias, bias};
    return __builtin_convertvector(acc + bv, bf16x8);
}
__device__ __forceinline__ bf16x8 bias_relu_cvt(v8f acc, float bias) {
    v8f bv = {bias, bias, bias, bias, bias, bias, bias, bias};
    v8f z  = {};
    return __builtin_convertvector(__builtin_elementwise_max(acc + bv, z), bf16x8);
}

// ---------------- weight prep: fp32 -> bf16, transposed, into workspace --------
__global__ void prep_weights(const float* __restrict__ W_in, const float* __restrict__ W_h,
                             bf16_t* __restrict__ WhT, bf16_t* __restrict__ WinT) {
    const int WH_TOTAL = HIDDEN * UNITS * UNITS;
    int idx = blockIdx.x * blockDim.x + threadIdx.x;
    if (idx < WH_TOTAL) {
        int l   = idx / (UNITS * UNITS);
        int rem = idx % (UNITS * UNITS);
        int o   = rem / UNITS;   // output unit (row of transposed)
        int i   = rem % UNITS;   // input unit
        WhT[idx] = (bf16_t)W_h[l * UNITS * UNITS + i * UNITS + o];
    } else if (idx < WH_TOTAL + UNITS * KIN) {
        int rem = idx - WH_TOTAL;
        int o = rem / KIN;
        int i = rem % KIN;
        WinT[rem] = (i < 16) ? (bf16_t)W_in[i * UNITS + o] : (bf16_t)0.f;
    }
}

// ---------------- main fused GCN kernel: one workgroup per graph ---------------
__global__ void __launch_bounds__(256)
gcn_fused_kernel(const int*   __restrict__ pdg,
                 const float* __restrict__ feat,
                 const float* __restrict__ adj,
                 const float* __restrict__ mask,
                 const float* __restrict__ emb_table,
                 const float* __restrict__ b_in,
                 const float* __restrict__ b_h,
                 const float* __restrict__ W_out,
                 const float* __restrict__ b_out,
                 const bf16_t* __restrict__ wsWhT,
                 const bf16_t* __restrict__ wsWinT,
                 float* __restrict__ out) {
    __shared__ __align__(32) bf16_t adjS[NN * NN];       // 32 KB, A of adjacency GEMM
    __shared__ __align__(32) bf16_t xA[NN * UNITS];      // 32 KB, node-major activations (A op)
    __shared__ __align__(32) bf16_t hT[UNITS * NN];      // 32 KB, unit-major hidden (B op); aliased as xin
    __shared__ __align__(32) bf16_t WT[UNITS * UNITS];   // 32 KB, W^T (B op); aliased as winT
    __shared__ float biasS[UNITS];
    __shared__ float maskS[NN];
    __shared__ float red[UNITS];

    const int tid  = threadIdx.x;
    const int lane = tid & 31;
    const int w    = tid >> 5;            // wave id 0..7 = C-tile row
    const int b    = blockIdx.x;

    // ---- stage adj (fp32 global -> bf16 LDS), coalesced float4, packed cvt ----
    {
        const v4f* src = (const v4f*)(adj + (size_t)b * NN * NN);
        for (int i = tid; i < (NN * NN) / 4; i += 256) {
            v4f v = src[i];
            *(bf16x4*)&adjS[i * 4] = __builtin_convertvector(v, bf16x4);
        }
    }
    // ---- build padded input rows: [feat(8) | emb(8) | zeros(16)] ----
    bf16_t* xin = hT;   // alias (only used before layer 0 writes hT)
    if (tid < NN) {
        const float* f = feat + ((size_t)b * NN + tid) * 8;
        int id = pdg[(size_t)b * NN + tid];
        const float* e = emb_table + id * 8;
        bf16_t* row = xin + tid * KIN;
        *(bf16x4*)(row +  0) = __builtin_convertvector(*(const v4f*)(f + 0), bf16x4);
        *(bf16x4*)(row +  4) = __builtin_convertvector(*(const v4f*)(f + 4), bf16x4);
        *(bf16x4*)(row +  8) = __builtin_convertvector(*(const v4f*)(e + 0), bf16x4);
        *(bf16x4*)(row + 12) = __builtin_convertvector(*(const v4f*)(e + 4), bf16x4);
        *(bf16x8*)(row + 16) = (bf16x8){};
        *(bf16x8*)(row + 24) = (bf16x8){};
        maskS[tid] = mask[(size_t)b * NN + tid];
    }
    // ---- stage W_in^T (padded) and b_in ----
    bf16_t* winT = WT;  // alias
    {
        const uint4* s = (const uint4*)wsWinT;
        uint4* d = (uint4*)winT;
        for (int i = tid; i < (UNITS * KIN) / 8; i += 256) d[i] = s[i];
    }
    if (tid < UNITS) biasS[tid] = b_in[tid];
    __syncthreads();

    // ---- input layer: xA = xin @ W_in + b_in (K=32, single WMMA per tile) ----
    {
        v16bf a = load_a_frag(xin + 16 * w * KIN, KIN);
        const int n = lane & 15, mb = (lane >> 4) << 3;
#pragma unroll
        for (int nt = 0; nt < 8; ++nt) {
            v16bf bb = load_bt_frag(winT + (16 * nt) * KIN, KIN);
            v8f acc = {};
            acc = wmma_bf16(a, bb, acc);
            const int u = 16 * nt + n;
            const int m0 = 16 * w + mb;
            bf16x8 o = bias_cvt(acc, biasS[u]);
#pragma unroll
            for (int r = 0; r < 8; ++r)
                xA[(m0 + r) * UNITS + u] = o[r];
        }
    }

    // ---- 6 hidden layers: h = relu(x@W + b);  x = adj @ h ----
    for (int l = 0; l < HIDDEN; ++l) {
        __syncthreads();
        {   // stage layer weights (bf16, pre-transposed) + bias
            const uint4* s = (const uint4*)(wsWhT + l * UNITS * UNITS);
            uint4* d = (uint4*)WT;
            for (int i = tid; i < (UNITS * UNITS) / 8; i += 256) d[i] = s[i];
        }
        if (tid < UNITS) biasS[tid] = b_h[l * UNITS + tid];
        __syncthreads();

        // GEMM1: hT[u][m] = relu( sum_k xA[m][k] * W[k][u] + b[u] )
        {
            const bf16_t* arow = xA + (16 * w) * UNITS;
            v16bf a0 = load_a_frag(arow +  0, UNITS);
            v16bf a1 = load_a_frag(arow + 32, UNITS);
            v16bf a2 = load_a_frag(arow + 64, UNITS);
            v16bf a3 = load_a_frag(arow + 96, UNITS);
            const int n = lane & 15, mb = (lane >> 4) << 3;
#pragma unroll
            for (int nt = 0; nt < 8; ++nt) {
                const bf16_t* bbase = WT + (16 * nt) * UNITS;
                v8f acc = {};
                acc = wmma_bf16(a0, load_bt_frag(bbase +  0, UNITS), acc);
                acc = wmma_bf16(a1, load_bt_frag(bbase + 32, UNITS), acc);
                acc = wmma_bf16(a2, load_bt_frag(bbase + 64, UNITS), acc);
                acc = wmma_bf16(a3, load_bt_frag(bbase + 96, UNITS), acc);
                const int u = 16 * nt + n;
                bf16x8 o = bias_relu_cvt(acc, biasS[u]);
                // transposed store: one contiguous 16B store per lane
                *(bf16x8*)&hT[u * NN + 16 * w + mb] = o;
            }
        }
        __syncthreads();

        // GEMM2: xA[i][u] = sum_j adj[i][j] * h[j][u]   (B op = hT, already transposed)
        {
            const bf16_t* arow = adjS + (16 * w) * NN;
            v16bf a0 = load_a_frag(arow +  0, NN);
            v16bf a1 = load_a_frag(arow + 32, NN);
            v16bf a2 = load_a_frag(arow + 64, NN);
            v16bf a3 = load_a_frag(arow + 96, NN);
            const int n = lane & 15, mb = (lane >> 4) << 3;
#pragma unroll
            for (int nt = 0; nt < 8; ++nt) {
                const bf16_t* bbase = hT + (16 * nt) * NN;
                v8f acc = {};
                acc = wmma_bf16(a0, load_bt_frag(bbase +  0, NN), acc);
                acc = wmma_bf16(a1, load_bt_frag(bbase + 32, NN), acc);
                acc = wmma_bf16(a2, load_bt_frag(bbase + 64, NN), acc);
                acc = wmma_bf16(a3, load_bt_frag(bbase + 96, NN), acc);
                const int u = 16 * nt + n;
                const int m0 = 16 * w + mb;
                bf16x8 o = __builtin_convertvector(acc, bf16x8);
#pragma unroll
                for (int r = 0; r < 8; ++r)
                    xA[(m0 + r) * UNITS + u] = o[r];
            }
        }
    }
    __syncthreads();

    // ---- masked mean pooling + output head ----
    if (tid < UNITS) {
        float s = 0.f;
#pragma unroll 4
        for (int n = 0; n < NN; ++n)
            s += (float)xA[n * UNITS + tid] * maskS[n];
        red[tid] = s * W_out[tid];      // W_out is [UNITS][1]
    }
    __syncthreads();
    if (tid == 0) {
        float denom = 0.f, t = 0.f;
        for (int n = 0; n < NN; ++n) denom += maskS[n];
        if (denom < 1.f) denom = 1.f;
        for (int u = 0; u < UNITS; ++u) t += red[u];
        out[b] = t / denom + b_out[0];
    }
}

extern "C" void kernel_launch(void* const* d_in, const int* in_sizes, int n_in,
                              void* d_out, int out_size, void* d_ws, size_t ws_size,
                              hipStream_t stream) {
    const int*   pdg      = (const int*)  d_in[0];
    const float* feat     = (const float*)d_in[1];
    const float* adj      = (const float*)d_in[2];
    const float* mask     = (const float*)d_in[3];
    const float* embt     = (const float*)d_in[4];
    const float* W_in     = (const float*)d_in[5];
    const float* b_in     = (const float*)d_in[6];
    const float* W_h      = (const float*)d_in[7];
    const float* b_h      = (const float*)d_in[8];
    const float* W_out    = (const float*)d_in[9];
    const float* b_out    = (const float*)d_in[10];

    const int B = in_sizes[0] / NN;   // pdg is [B, N]

    bf16_t* WhT  = (bf16_t*)d_ws;
    bf16_t* WinT = WhT + HIDDEN * UNITS * UNITS;

    const int prep_total = HIDDEN * UNITS * UNITS + UNITS * KIN;
    prep_weights<<<(prep_total + 255) / 256, 256, 0, stream>>>(W_in, W_h, WhT, WinT);

    gcn_fused_kernel<<<B, 256, 0, stream>>>(pdg, feat, adj, mask, embt,
                                            b_in, b_h, W_out, b_out,
                                            WhT, WinT, (float*)d_out);
}
